// DecoderBlock_63273458205354
// MI455X (gfx1250) — matrix operand order repaired and checked by reading, not compiled, not run on hardware
//
#include <hip/hip_runtime.h>
#include <hip/hip_bf16.h>
#include <math.h>

// ---------------------------------------------------------------------------
// Types for CDNA5 WMMA (wave32): bf16 A/B fragments (16 halves = 8 VGPRs),
// f32 C/D fragments (8 VGPRs).
// ---------------------------------------------------------------------------
typedef __attribute__((ext_vector_type(16))) __bf16 v16bf;
typedef __attribute__((ext_vector_type(8)))  float  v8f;

__device__ __forceinline__ unsigned short f2bf(float f) {
    unsigned int u = __float_as_uint(f);
    unsigned int r = u + 0x7FFFu + ((u >> 16) & 1u);   // round-to-nearest-even
    return (unsigned short)(r >> 16);
}

__device__ __forceinline__ v16bf make_frag(uint4 lo, uint4 hi) {
    union U { uint4 u[2]; v16bf v; } c;
    c.u[0] = lo; c.u[1] = hi;
    return c.v;
}

__device__ __forceinline__ v8f wmma_bf16(v16bf a, v16bf b, v8f c) {
    // D = A(16x32) * B(32x16) + C, f32 accumulate
    return __builtin_amdgcn_wmma_f32_16x16x32_bf16(
        false, a, false, b, (short)0, c, false, false);
}

// xor-butterfly reductions within 16-lane groups via DPP ROW_XMASK
// (VALU-only: avoids ds_bpermute that __shfl_xor lowers to)
__device__ __forceinline__ float xmax16(float x) {
    x = fmaxf(x, __int_as_float(__builtin_amdgcn_update_dpp(
            0, __float_as_int(x), 0x161, 0xf, 0xf, true)));
    x = fmaxf(x, __int_as_float(__builtin_amdgcn_update_dpp(
            0, __float_as_int(x), 0x162, 0xf, 0xf, true)));
    x = fmaxf(x, __int_as_float(__builtin_amdgcn_update_dpp(
            0, __float_as_int(x), 0x164, 0xf, 0xf, true)));
    x = fmaxf(x, __int_as_float(__builtin_amdgcn_update_dpp(
            0, __float_as_int(x), 0x168, 0xf, 0xf, true)));
    return x;
}
__device__ __forceinline__ float xsum16(float x) {
    x += __int_as_float(__builtin_amdgcn_update_dpp(
            0, __float_as_int(x), 0x161, 0xf, 0xf, true));
    x += __int_as_float(__builtin_amdgcn_update_dpp(
            0, __float_as_int(x), 0x162, 0xf, 0xf, true));
    x += __int_as_float(__builtin_amdgcn_update_dpp(
            0, __float_as_int(x), 0x164, 0xf, 0xf, true));
    x += __int_as_float(__builtin_amdgcn_update_dpp(
            0, __float_as_int(x), 0x168, 0xf, 0xf, true));
    return x;
}

// ---------------------------------------------------------------------------
// fp32 -> bf16 weight conversion
// ---------------------------------------------------------------------------
__global__ void cvt_f32_bf16(const float* __restrict__ in,
                             unsigned short* __restrict__ out, int n) {
    int i = blockIdx.x * blockDim.x + threadIdx.x;
    if (i < n) out[i] = f2bf(in[i]);
}

// ---------------------------------------------------------------------------
// LayerNorm over rows of 1024, output bf16.  One 256-thread block per row.
// ---------------------------------------------------------------------------
__global__ __launch_bounds__(256)
void layernorm_bf16(const float* __restrict__ x,
                    const float* __restrict__ g,
                    const float* __restrict__ b,
                    unsigned short* __restrict__ out) {
    const int row = blockIdx.x;
    const int tid = threadIdx.x;
    const float* xr = x + (size_t)row * 1024;

    float v0 = xr[tid], v1 = xr[tid + 256], v2 = xr[tid + 512], v3 = xr[tid + 768];
    float s  = v0 + v1 + v2 + v3;
    float s2 = v0 * v0 + v1 * v1 + v2 * v2 + v3 * v3;
    #pragma unroll
    for (int m = 1; m < 32; m <<= 1) {
        s  += __shfl_xor(s,  m);
        s2 += __shfl_xor(s2, m);
    }
    __shared__ float ss[8], ss2[8];
    if ((tid & 31) == 0) { ss[tid >> 5] = s; ss2[tid >> 5] = s2; }
    __syncthreads();
    float tot = 0.f, tot2 = 0.f;
    #pragma unroll
    for (int i = 0; i < 8; ++i) { tot += ss[i]; tot2 += ss2[i]; }
    const float mu   = tot * (1.0f / 1024.0f);
    const float var  = tot2 * (1.0f / 1024.0f) - mu * mu;
    const float rstd = rsqrtf(var + 1e-5f);

    unsigned short* orow = out + (size_t)row * 1024;
    orow[tid]       = f2bf((v0 - mu) * rstd * g[tid]       + b[tid]);
    orow[tid + 256] = f2bf((v1 - mu) * rstd * g[tid + 256] + b[tid + 256]);
    orow[tid + 512] = f2bf((v2 - mu) * rstd * g[tid + 512] + b[tid + 512]);
    orow[tid + 768] = f2bf((v3 - mu) * rstd * g[tid + 768] + b[tid + 768]);
}

// ---------------------------------------------------------------------------
// Tiled bf16 WMMA GEMM:  C[M,N] = A[M,K] * W[K,N]  (+ bias, epilogue modes)
//   MODE 0: out_bf16 = acc + bias              (QKV)
//   MODE 1: out_f32  = acc + bias + resid      (O-proj / final proj)
//   MODE 2: out_bf16 = gelu(acc + bias)        (FC)
// Block: 256 threads (8 waves), tile 128x128, K-step 32.
// Wave grid 2(M) x 4(N); each wave: 4x2 grid of 16x16 WMMA tiles.
// ---------------------------------------------------------------------------
#define LDSA 40   // A tile padded row stride (ushort) for 32-wide K
#define LDSB 20   // B tile padded row stride (uint, K-pairs)

template <int MODE>
__global__ __launch_bounds__(256)
void gemm_bf16(const unsigned short* __restrict__ A, int lda,
               const unsigned short* __restrict__ Bw, int ldb,
               const float* __restrict__ bias,
               const float* __restrict__ resid,
               float* __restrict__ outF,
               unsigned short* __restrict__ outH,
               int M, int K) {
    __shared__ unsigned short As[128 * LDSA];
    __shared__ unsigned int   Bs[128 * LDSB];   // transposed, K packed in pairs

    const int tid  = threadIdx.x;
    const int lane = tid & 31;
    const int wv   = tid >> 5;
    const int mw   = wv >> 2;        // 0..1
    const int nw   = wv & 3;         // 0..3
    const int m0   = blockIdx.y * 128;
    const int n0   = blockIdx.x * 128;
    const int N    = ldb;

    v8f acc[4][2];
    #pragma unroll
    for (int im = 0; im < 4; ++im)
        #pragma unroll
        for (int in = 0; in < 2; ++in)
            acc[im][in] = (v8f){0.f,0.f,0.f,0.f,0.f,0.f,0.f,0.f};

    // A staging: 128 rows x 32 k; thread -> (row, 16-k chunk)
    const int arow = tid >> 1, acol = (tid & 1) * 16;
    // B staging: k-pair p = tid>>4 (0..15), 8 columns nb
    const int bp = tid >> 4, bn = (tid & 15) * 8;

    const int kb  = (lane >> 4) * 8;    // A-frag K chunk base (ushort)
    const int kp2 = (lane >> 4) * 4;    // B-frag K chunk base (uint pairs)

    const unsigned short* apG = A + (size_t)(m0 + arow) * lda + acol;
    const unsigned short* bpG = Bw + (size_t)(2 * bp) * ldb + n0 + bn;

    for (int k0 = 0; k0 < K; k0 += 32) {
        __syncthreads();
        // stage A tile (row-major)
        {
            const unsigned short* ap = apG + k0;
            uint4 a0 = *(const uint4*)ap;
            uint4 a1 = *(const uint4*)(ap + 8);
            *(uint4*)&As[arow * LDSA + acol]     = a0;
            *(uint4*)&As[arow * LDSA + acol + 8] = a1;
        }
        // stage B tile transposed with K-pair packing:
        // Bs[n*LDSB + p] = W[k0+2p][n] | (W[k0+2p+1][n] << 16)
        {
            const unsigned short* b0p = bpG + (size_t)k0 * ldb;
            uint4 r0 = *(const uint4*)b0p;
            uint4 r1 = *(const uint4*)(b0p + ldb);
            unsigned short t0[8], t1[8];
            *(uint4*)t0 = r0; *(uint4*)t1 = r1;
            #pragma unroll
            for (int i = 0; i < 8; ++i)
                Bs[(bn + i) * LDSB + bp] =
                    (unsigned int)t0[i] | ((unsigned int)t1[i] << 16);
        }
        // prefetch next K-step tiles into cache while this step computes
        if (k0 + 32 < K) {
            __builtin_prefetch(apG + k0 + 32, 0, 3);
            __builtin_prefetch(bpG + (size_t)(k0 + 32) * ldb, 0, 3);
        }
        __syncthreads();

        // load fragments
        v16bf af[4];
        #pragma unroll
        for (int im = 0; im < 4; ++im) {
            const int r = mw * 64 + im * 16 + (lane & 15);
            af[im] = make_frag(*(const uint4*)&As[r * LDSA + kb],
                               *(const uint4*)&As[r * LDSA + 16 + kb]);
        }
        v16bf bfr[2];
        #pragma unroll
        for (int in = 0; in < 2; ++in) {
            const int c = nw * 32 + in * 16 + (lane & 15);
            const unsigned int* br = &Bs[c * LDSB + kp2];
            bfr[in] = make_frag(*(const uint4*)br, *(const uint4*)(br + 8));
        }
        #pragma unroll
        for (int im = 0; im < 4; ++im)
            #pragma unroll
            for (int in = 0; in < 2; ++in)
                acc[im][in] = wmma_bf16(af[im], bfr[in], acc[im][in]);
    }

    // ---- straight-line epilogue (templated, no runtime branching) ----
    #pragma unroll
    for (int im = 0; im < 4; ++im) {
        #pragma unroll
        for (int in = 0; in < 2; ++in) {
            const int colg = n0 + nw * 32 + in * 16 + (lane & 15);
            const int rowb = m0 + mw * 64 + im * 16 + 8 * (lane >> 4);
            const float bv = bias[colg];
            const size_t base = (size_t)rowb * N + colg;
            if (MODE == 1) {
                float*       po = outF  + base;
                const float* pr = resid + base;
                #pragma unroll
                for (int r = 0; r < 8; ++r) {
                    po[(size_t)r * N] = acc[im][in][r] + bv + pr[(size_t)r * N];
                }
            } else if (MODE == 0) {
                unsigned short* po = outH + base;
                #pragma unroll
                for (int r = 0; r < 8; ++r)
                    po[(size_t)r * N] = f2bf(acc[im][in][r] + bv);
            } else {
                unsigned short* po = outH + base;
                #pragma unroll
                for (int r = 0; r < 8; ++r) {
                    float v = acc[im][in][r] + bv;
                    po[(size_t)r * N] =
                        f2bf(0.5f * v * (1.0f + erff(v * 0.70710678118f)));
                }
            }
        }
    }
}

// ---------------------------------------------------------------------------
// Causal flash attention.  One block = 8 waves = 8 consecutive 16-query
// blocks of the SAME (b, h).  K and V^T tiles for each 64-key step are
// staged cooperatively once per block and shared by all 8 waves.  64-key
// steps halve the per-key cost of softmax reductions / O-rescale / barriers.
// qkv: [4096][3072] bf16 (q | k | v), head dim 64, scale 1/8.
// out attn: [4096][1024] bf16.
// ---------------------------------------------------------------------------
__global__ __launch_bounds__(256)
void attention_bf16(const unsigned short* __restrict__ qkv,
                    unsigned short* __restrict__ attn) {
    __shared__ unsigned short Kt[64 * 64];      // K tile, row-major [k][d]
    __shared__ unsigned short Vt[64 * 64];      // V^T tile [d][k]
    __shared__ unsigned short Pb[8][16 * 64];   // per-wave P tile (16x64)

    const int tid  = threadIdx.x;
    const int lane = tid & 31;
    const int wv   = tid >> 5;
    const int task0 = blockIdx.x * 8;            // first task of block
    const int qb0  = task0 & 127;                // aligned to 8
    const int h    = (task0 >> 7) & 15;
    const int b    = task0 >> 11;
    const int qbase = (qb0 + wv) * 16;           // this wave's query block
    const int qlast = qbase + 15;

    unsigned short* P = Pb[wv];

    const int kb  = (lane >> 4) * 8;    // A-frag chunk base
    const int kbB = (lane >> 4) * 8;    // B-frag chunk base

    // staging assignment: thread -> (key row, 16-wide d chunk)
    const int vrow = tid >> 2;          // 0..63
    const int dc   = (tid & 3) * 16;    // 0,16,32,48

    // Q fragments (A layout): two K=32 halves of the 16x64 Q tile
    v16bf qa0, qa1;
    {
        const unsigned short* qp =
            qkv + (size_t)(b * 2048 + qbase + (lane & 15)) * 3072 + h * 64;
        qa0 = make_frag(*(const uint4*)(qp + kb),      *(const uint4*)(qp + 16 + kb));
        qa1 = make_frag(*(const uint4*)(qp + 32 + kb), *(const uint4*)(qp + 48 + kb));
    }

    v8f o[4];
    #pragma unroll
    for (int dt = 0; dt < 4; ++dt)
        o[dt] = (v8f){0.f,0.f,0.f,0.f,0.f,0.f,0.f,0.f};
    float mS[8], lS[8];
    #pragma unroll
    for (int r = 0; r < 8; ++r) { mS[r] = -INFINITY; lS[r] = 0.f; }

    const int qrow = qbase + 8 * (lane >> 4);      // + r gives this lane's row
    const int nblk = ((qb0 * 16 + 127) >> 6) + 1;  // 64-key blocks for block

    for (int j = 0; j < nblk; ++j) {
        const int keybase = j * 64;

        __syncthreads();   // previous iteration's tile reads complete
        // ---- cooperative staging: K tile (row-major) and V^T tile
        {
            const size_t rb = (size_t)(b * 2048 + keybase + vrow) * 3072 + h * 64 + dc;
            uint4 ka = *(const uint4*)(qkv + rb + 1024);
            uint4 kc = *(const uint4*)(qkv + rb + 1024 + 8);
            *(uint4*)&Kt[vrow * 64 + dc]     = ka;
            *(uint4*)&Kt[vrow * 64 + dc + 8] = kc;
            uint4 va = *(const uint4*)(qkv + rb + 2048);
            uint4 vc = *(const uint4*)(qkv + rb + 2048 + 8);
            unsigned short tv[16];
            *(uint4*)&tv[0] = va; *(uint4*)&tv[8] = vc;
            #pragma unroll
            for (int i = 0; i < 16; ++i)
                Vt[(dc + i) * 64 + vrow] = tv[i];
        }
        __syncthreads();

        if (keybase > qlast) continue;   // causally done for this wave

        // ---- scores S = Q K^T * 1/8, four 16-key n-tiles (K from LDS)
        v8f s[4];
        #pragma unroll
        for (int nt = 0; nt < 4; ++nt) {
            const unsigned short* kp = &Kt[(nt * 16 + (lane & 15)) * 64 + kbB];
            v16bf kf0 = make_frag(*(const uint4*)kp,        *(const uint4*)(kp + 16));
            v16bf kf1 = make_frag(*(const uint4*)(kp + 32), *(const uint4*)(kp + 48));
            v8f t = (v8f){0.f,0.f,0.f,0.f,0.f,0.f,0.f,0.f};
            t = wmma_bf16(qa0, kf0, t);
            t = wmma_bf16(qa1, kf1, t);
            s[nt] = t;
        }

        // ---- scale + causal mask
        const int col0 = keybase + (lane & 15);
        #pragma unroll
        for (int nt = 0; nt < 4; ++nt) {
            const int key = col0 + nt * 16;
            #pragma unroll
            for (int r = 0; r < 8; ++r) {
                float v = s[nt][r] * 0.125f;
                s[nt][r] = (key <= qrow + r) ? v : -INFINITY;
            }
        }

        // ---- online softmax update (DPP row reductions, VALU-only)
        #pragma unroll
        for (int r = 0; r < 8; ++r) {
            float rm = fmaxf(fmaxf(s[0][r], s[1][r]), fmaxf(s[2][r], s[3][r]));
            rm = xmax16(rm);
            const float mnew  = fmaxf(mS[r], rm);
            const float alpha = __expf(mS[r] - mnew);
            float rs = 0.f;
            #pragma unroll
            for (int nt = 0; nt < 4; ++nt) {
                float p = __expf(s[nt][r] - mnew);
                s[nt][r] = p;
                rs += p;
            }
            rs = xsum16(rs);
            lS[r] = lS[r] * alpha + rs;
            mS[r] = mnew;
            #pragma unroll
            for (int dt = 0; dt < 4; ++dt) o[dt][r] *= alpha;
        }

        // ---- write P (C layout -> LDS row-major, bf16)
        {
            const int Mb = 8 * (lane >> 4);
            const int c  = lane & 15;
            #pragma unroll
            for (int r = 0; r < 8; ++r)
                #pragma unroll
                for (int nt = 0; nt < 4; ++nt)
                    P[(Mb + r) * 64 + c + nt * 16] = f2bf(s[nt][r]);
        }
        asm volatile("s_wait_dscnt 0" ::: "memory");   // intra-wave LDS RAW

        // ---- P fragments (A 16x64 = two K=32 halves)
        v16bf pa0, pa1;
        {
            const unsigned short* pp = P + (lane & 15) * 64;
            pa0 = make_frag(*(const uint4*)(pp + kb),      *(const uint4*)(pp + 16 + kb));
            pa1 = make_frag(*(const uint4*)(pp + 32 + kb), *(const uint4*)(pp + 48 + kb));
        }
        // ---- O += P @ V  (4 d-tiles of 16, chained over K=64)
        #pragma unroll
        for (int dt = 0; dt < 4; ++dt) {
            const unsigned short* vt = &Vt[(dt * 16 + (lane & 15)) * 64 + kbB];
            v16bf vb0 = make_frag(*(const uint4*)vt,        *(const uint4*)(vt + 16));
            v16bf vb1 = make_frag(*(const uint4*)(vt + 32), *(const uint4*)(vt + 48));
            o[dt] = wmma_bf16(pa0, vb0, o[dt]);
            o[dt] = wmma_bf16(pa1, vb1, o[dt]);
        }
    }

    // ---- normalize and store
    unsigned short* op = attn + (size_t)(b * 2048 + qbase) * 1024 + h * 64;
    const int Mb = 8 * (lane >> 4);
    const int c  = lane & 15;
    #pragma unroll
    for (int r = 0; r < 8; ++r) {
        const float inv = 1.0f / lS[r];
        #pragma unroll
        for (int dt = 0; dt < 4; ++dt)
            op[(size_t)(Mb + r) * 1024 + dt * 16 + c] = f2bf(o[dt][r] * inv);
    }
}

// ---------------------------------------------------------------------------
// Host launcher
// ---------------------------------------------------------------------------
extern "C" void kernel_launch(void* const* d_in, const int* in_sizes, int n_in,
                              void* d_out, int out_size, void* d_ws, size_t ws_size,
                              hipStream_t stream) {
    const float* x     = (const float*)d_in[0];
    const float* ln1_g = (const float*)d_in[1];
    const float* ln1_b = (const float*)d_in[2];
    const float* ln2_g = (const float*)d_in[3];
    const float* ln2_b = (const float*)d_in[4];
    const float* W_qkv = (const float*)d_in[5];
    const float* b_qkv = (const float*)d_in[6];
    const float* W_o   = (const float*)d_in[7];
    const float* b_o   = (const float*)d_in[8];
    const float* W_fc  = (const float*)d_in[9];
    const float* b_fc  = (const float*)d_in[10];
    const float* W_pr  = (const float*)d_in[11];
    const float* b_pr  = (const float*)d_in[12];
    float* out = (float*)d_out;

    const int M = 4096;   // B*T
    char* ws = (char*)d_ws;
    auto alloc = [&](size_t bytes) {
        char* p = ws;
        ws += (bytes + 255) & ~(size_t)255;
        return p;
    };
    unsigned short* wq   = (unsigned short*)alloc((size_t)1024 * 3072 * 2);
    unsigned short* wo   = (unsigned short*)alloc((size_t)1024 * 1024 * 2);
    unsigned short* wfc  = (unsigned short*)alloc((size_t)1024 * 4096 * 2);
    unsigned short* wpr  = (unsigned short*)alloc((size_t)4096 * 1024 * 2);
    unsigned short* h1   = (unsigned short*)alloc((size_t)M * 1024 * 2);
    unsigned short* qkvb = (unsigned short*)alloc((size_t)M * 3072 * 2);
    unsigned short* attb = (unsigned short*)alloc((size_t)M * 1024 * 2);
    float*          xres = (float*)        alloc((size_t)M * 1024 * 4);
    unsigned short* h2   = (unsigned short*)alloc((size_t)M * 1024 * 2);
    unsigned short* act  = (unsigned short*)alloc((size_t)M * 4096 * 2);

    // weight conversion
    cvt_f32_bf16<<<(3145728 + 255) / 256, 256, 0, stream>>>(W_qkv, wq, 3145728);
    cvt_f32_bf16<<<(1048576 + 255) / 256, 256, 0, stream>>>(W_o,   wo, 1048576);
    cvt_f32_bf16<<<(4194304 + 255) / 256, 256, 0, stream>>>(W_fc,  wfc, 4194304);
    cvt_f32_bf16<<<(4194304 + 255) / 256, 256, 0, stream>>>(W_pr,  wpr, 4194304);

    // LN1
    layernorm_bf16<<<M, 256, 0, stream>>>(x, ln1_g, ln1_b, h1);
    // QKV = h1 @ W_qkv + b  -> bf16
    gemm_bf16<0><<<dim3(3072 / 128, M / 128), 256, 0, stream>>>(
        h1, 1024, wq, 3072, b_qkv, nullptr, nullptr, qkvb, M, 1024);
    // attention
    attention_bf16<<<512, 256, 0, stream>>>(qkvb, attb);
    // xres = x + attn @ W_o + b_o   (f32)
    gemm_bf16<1><<<dim3(1024 / 128, M / 128), 256, 0, stream>>>(
        attb, 1024, wo, 1024, b_o, x, xres, nullptr, M, 1024);
    // LN2
    layernorm_bf16<<<M, 256, 0, stream>>>(xres, ln2_g, ln2_b, h2);
    // act = gelu(h2 @ W_fc + b_fc) -> bf16
    gemm_bf16<2><<<dim3(4096 / 128, M / 128), 256, 0, stream>>>(
        h2, 1024, wfc, 4096, b_fc, nullptr, nullptr, act, M, 1024);
    // out = xres + act @ W_proj + b_proj (f32)
    gemm_bf16<1><<<dim3(1024 / 128, M / 128), 256, 0, stream>>>(
        act, 4096, wpr, 1024, b_pr, xres, out, nullptr, M, 4096);
}
